// NT_Xent_Loss_61624190763370
// MI455X (gfx1250) — compile-verified, matrix operand order checked by reading, not run on hardware
//
#include <hip/hip_runtime.h>

// NT-Xent loss on MI455X (gfx1250), fused and WMMA-based.
//   N = 8192 rows, D = 256, TEMP = 0.5  ->  sim/T = 2*cos in [-2, 2].
// Fixed-shift LSE: lse = 2 + log(sum_{j!=i} exp(sim_ij/T - 2)) -- no online max,
// so the masked row-logsumexp is an order-free sum fused into the WMMA tile loop.
//
// Dataflow: block = 8 waves, each wave owns a 16-row stripe (128 rows/block);
// all waves walk the SAME column-tile sequence so the 8KB B tile gets 8x reuse
// from WGP$ (no LDS staging, no barriers). Grid is 64 row-blocks x 8 col-groups;
// per-group partial exp-sums land in partialS[row][8], combined deterministically
// by a finalize kernel.

#define BATCH_ 4096
#define N_TOT  8192
#define DIM_   256
#define NTILE  (N_TOT / 16)      // 512 column tiles
#define CGRP   8                 // column groups
#define TPG    (NTILE / CGRP)    // 64 tiles per group

typedef __attribute__((ext_vector_type(16))) _Float16 v16h;
typedef __attribute__((ext_vector_type(8)))  _Float16 v8h;
typedef __attribute__((ext_vector_type(8)))  float    v8f;

// ---------------------------------------------------------------------------
// Kernel 1: row normalization (fp32 -> unit-norm f16). One wave per row.
// ---------------------------------------------------------------------------
__global__ __launch_bounds__(256) void normalize_rows(
    const float* __restrict__ zi, const float* __restrict__ zj,
    _Float16* __restrict__ zn)
{
    const int row  = blockIdx.x * 8 + (threadIdx.x >> 5);
    const int lane = threadIdx.x & 31;
    const float* src = (row < BATCH_) ? (zi + row * DIM_)
                                      : (zj + (row - BATCH_) * DIM_);
    const float4* s4 = (const float4*)(src + lane * 8);
    float4 a = s4[0], b = s4[1];
    float ss = a.x*a.x + a.y*a.y + a.z*a.z + a.w*a.w
             + b.x*b.x + b.y*b.y + b.z*b.z + b.w*b.w;
#pragma unroll
    for (int off = 16; off > 0; off >>= 1)
        ss += __shfl_xor(ss, off, 32);
    const float inv = 1.0f / fmaxf(sqrtf(ss), 1e-8f);

    v8h o;
    o[0] = (_Float16)(a.x * inv); o[1] = (_Float16)(a.y * inv);
    o[2] = (_Float16)(a.z * inv); o[3] = (_Float16)(a.w * inv);
    o[4] = (_Float16)(b.x * inv); o[5] = (_Float16)(b.y * inv);
    o[6] = (_Float16)(b.z * inv); o[7] = (_Float16)(b.w * inv);
    *(v8h*)(zn + row * DIM_ + lane * 8) = o;
}

// ---------------------------------------------------------------------------
// CDNA5 16-bit A/B fragment layout: lane = half*16 + idx owns row/col base+idx;
// elems 0..7 = K[k0 + half*8 ..], elems 8..15 = K[k0 + 16 + half*8 ..].
// ---------------------------------------------------------------------------
__device__ __forceinline__ v16h load_frag(const _Float16* __restrict__ rowp,
                                          int k0, int half)
{
    v8h lo = *(const v8h*)(rowp + k0 + half * 8);
    v8h hi = *(const v8h*)(rowp + k0 + 16 + half * 8);
    v16h f;
#pragma unroll
    for (int e = 0; e < 8; ++e) { f[e] = lo[e]; f[e + 8] = hi[e]; }
    return f;
}

__device__ __forceinline__ void load_btile(v16h B[8], const _Float16* __restrict__ zn,
                                           int colBase, int idx, int half)
{
    const _Float16* brow = zn + (colBase + idx) * DIM_;
#pragma unroll
    for (int c = 0; c < 8; ++c) B[c] = load_frag(brow, c * 32, half);
}

__device__ __forceinline__ v8f tile_mm(const v16h A[8], const v16h B[8])
{
    v8f acc = {};
#pragma unroll
    for (int c = 0; c < 8; ++c)
        acc = __builtin_amdgcn_wmma_f32_16x16x32_f16(
            false, A[c], false, B[c], (short)0, acc, false, false);
    return acc;
}

// Post-process one 16x16 tile: shifted exp accumulation + register-only capture
// of the self-similarity term and the positive-pair logit (branch-free selects).
__device__ __forceinline__ void tile_post(const v8f& acc, float psum[8], int ctg,
                                          int diagTileW, int posTileW,
                                          bool ownDiag, int diagV,
                                          float& selfE, float& posX)
{
#pragma unroll
    for (int v = 0; v < 8; ++v) {
        float x = 2.0f * acc[v] - 2.0f;          // sim/T - 2
        float e = __expf(x);
        psum[v] += e;
        bool hit = ownDiag && (v == diagV);
        if ((ctg == diagTileW) && hit) selfE = e;
        if ((ctg == posTileW)  && hit) posX  = x + 2.0f;
    }
}

// ---------------------------------------------------------------------------
// Kernel 2: fused Gram tiles + masked shifted-exp partial row sums.
// grid = (64 row-blocks, 8 col-groups); block = 256 threads = 8 waves.
// Wave w owns rows R = (blockIdx.x*8 + w)*16 .. +16; processes 64 column tiles
// with register double-buffered B fragments (16-load clauses overlap 8 WMMAs).
// ---------------------------------------------------------------------------
__global__ __launch_bounds__(256, 1) void ntxent_main(
    const _Float16* __restrict__ zn,
    float* __restrict__ partialS,       // [N_TOT][CGRP]
    float* __restrict__ posArr)         // [N_TOT]
{
    const int tid  = threadIdx.x;
    const int wave = tid >> 5;
    const int lane = tid & 31;
    const int half = lane >> 4;
    const int idx  = lane & 15;

    const int R      = (blockIdx.x * 8 + wave) * 16;   // this wave's row stripe
    const int g      = blockIdx.y;                     // column group
    const int ctBase = g * TPG;                        // first tile index

    // A fragments: 16 rows x K=256, resident in 64 VGPRs, reused for 64 tiles.
    v16h A[8];
    {
        const _Float16* arow = zn + (R + idx) * DIM_;
#pragma unroll
        for (int c = 0; c < 8; ++c) A[c] = load_frag(arow, c * 32, half);
    }

    const int diagTileW = R >> 4;
    const int posTileW  = ((R + BATCH_) & (N_TOT - 1)) >> 4;
    const int diagV     = idx - half * 8;              // tile-diagonal slot
    const bool ownDiag  = (diagV >= 0) && (diagV < 8); // lane holds (m,m)

    float psum[8];
#pragma unroll
    for (int v = 0; v < 8; ++v) psum[v] = 0.0f;
    float selfE = 0.0f, posX = 0.0f;

    // Register double-buffered tile loop over this group's 64 column tiles.
    v16h B0[8], B1[8];
    load_btile(B0, zn, ctBase * 16, idx, half);
    for (int t = 0; t < TPG; t += 2) {
        load_btile(B1, zn, (ctBase + t + 1) * 16, idx, half);
        {
            v8f acc = tile_mm(A, B0);
            tile_post(acc, psum, ctBase + t, diagTileW, posTileW,
                      ownDiag, diagV, selfE, posX);
        }
        if (t + 2 < TPG) load_btile(B0, zn, (ctBase + t + 2) * 16, idx, half);
        {
            v8f acc = tile_mm(A, B1);
            tile_post(acc, psum, ctBase + t + 1, diagTileW, posTileW,
                      ownDiag, diagV, selfE, posX);
        }
    }

    // Remove the self-similarity contribution (exactly what was accumulated).
#pragma unroll
    for (int v = 0; v < 8; ++v)
        if (ownDiag && (v == diagV)) psum[v] -= selfE;

    // Fixed-order xor-shuffle reduction over the 16 column lanes of each half:
    // afterwards every lane holds full (group-partial) sums for rows half*8+v.
#pragma unroll
    for (int v = 0; v < 8; ++v) {
#pragma unroll
        for (int m = 1; m < 16; m <<= 1)
            psum[v] += __shfl_xor(psum[v], m, 32);
    }

    // One lane per half writes 8 row partials for this column group.
    if (idx == 0) {
#pragma unroll
        for (int v = 0; v < 8; ++v)
            partialS[(R + half * 8 + v) * CGRP + g] = psum[v];
    }

    // Positive-pair logit: written by exactly one block per row grid-wide.
    if (ownDiag && posTileW >= ctBase && posTileW < ctBase + TPG)
        posArr[R + idx] = posX;
}

// ---------------------------------------------------------------------------
// Kernel 3: deterministic finalize. S = fixed-order sum of 8 group partials,
// loss_row = 2 + log(S) - pos; fixed-order per-thread sums + fixed tree.
// ---------------------------------------------------------------------------
__global__ __launch_bounds__(256) void finalize(
    const float* __restrict__ partialS, const float* __restrict__ posArr,
    float* __restrict__ out)
{
    __shared__ float sm[256];
    const int t = threadIdx.x;
    float a = 0.0f;
    for (int r = t * 32; r < t * 32 + 32; ++r) {
        float S = 0.0f;
#pragma unroll
        for (int gi = 0; gi < CGRP; ++gi) S += partialS[r * CGRP + gi];
        a += 2.0f + __logf(S) - posArr[r];
    }
    sm[t] = a;
    __syncthreads();
#pragma unroll
    for (int off = 128; off > 0; off >>= 1) {
        if (t < off) sm[t] += sm[t + off];
        __syncthreads();
    }
    if (t == 0) out[0] = sm[0] / (float)N_TOT;
}

// ---------------------------------------------------------------------------
extern "C" void kernel_launch(void* const* d_in, const int* in_sizes, int n_in,
                              void* d_out, int out_size, void* d_ws, size_t ws_size,
                              hipStream_t stream)
{
    (void)in_sizes; (void)n_in; (void)out_size; (void)ws_size;
    const float* zi = (const float*)d_in[0];
    const float* zj = (const float*)d_in[1];

    char* ws = (char*)d_ws;
    _Float16* zn       = (_Float16*)ws;                                   // 4 MB
    float*    partialS = (float*)(ws + (size_t)N_TOT * DIM_ * 2);         // 256 KB
    float*    posArr   = (float*)(ws + (size_t)N_TOT * DIM_ * 2
                                     + (size_t)N_TOT * CGRP * 4);         // 32 KB

    normalize_rows<<<N_TOT / 8, 256, 0, stream>>>(zi, zj, zn);
    ntxent_main<<<dim3(N_TOT / 128, CGRP), 256, 0, stream>>>(zn, partialS, posArr);
    finalize<<<1, 256, 0, stream>>>(partialS, posArr, (float*)d_out);
}
